// CategoricalGraph_56418690400482
// MI455X (gfx1250) — matrix operand order repaired and checked by reading, not compiled, not run on hardware
//
#include <hip/hip_runtime.h>
#include <hip/hip_bf16.h>
#include <cstddef>

typedef __attribute__((ext_vector_type(16))) _Float16 v16h;
typedef __attribute__((ext_vector_type(8)))  float    v8f;

constexpr int NN = 20000;   // nodes
constexpr int TT = 20;      // time steps
constexpr int DD = 64;      // input dim
constexpr int HH = 64;      // hidden dim
constexpr int WW = 3;       // weeks
constexpr int EE = 640000;  // edges
constexpr int G3 = 3 * HH;  // 192 gate cols

// ---------------------------------------------------------------------------
// WMMA operand loaders (contiguous k = 16*hsel + s in both A and B, wave32).
// A: 16x32 f16 tile from f32 row-major source (converted on the fly).
// B: 32x16 f16 tile from pre-converted f16 weight, row-major (Nout, K);
//    one lane's 16 halves are contiguous -> single aligned 32B load.
// ---------------------------------------------------------------------------
__device__ inline v16h load_a_cvt(const float* __restrict__ arow, int k0,
                                  int hsel)
{
    v16h a;
#pragma unroll
    for (int v = 0; v < 8; ++v) {
        const int ka = k0 + 2 * v + hsel * 16;
        a[2 * v]     = (_Float16)arow[ka];
        a[2 * v + 1] = (_Float16)arow[ka + 1];
    }
    return a;
}

__device__ inline v16h load_b_f16(const _Float16* __restrict__ W, int ldw,
                                  int n_base, int k0, int hsel, int l15)
{
    // halves [k0 + hsel*16 .. +15] of weight row (n_base+l15): contiguous, 32B aligned
    return *(const v16h*)(W + (size_t)(n_base + l15) * ldw + k0 + hsel * 16);
}

// 2-step K=64 accumulate into one 16x16 f32 tile
__device__ inline v8f wmma_k64(const v16h a0, const v16h a1,
                               const v16h b0, const v16h b1)
{
    v8f acc = {};
    acc = __builtin_amdgcn_wmma_f32_16x16x32_f16(false, a0, false, b0,
                                                 (short)0, acc, false, false);
    acc = __builtin_amdgcn_wmma_f32_16x16x32_f16(false, a1, false, b1,
                                                 (short)0, acc, false, false);
    return acc;
}

// ---------------------------------------------------------------------------
// Weight f32 -> f16 pre-conversion (runs once, tiny)
// ---------------------------------------------------------------------------
__global__ void cvt_f16_kernel(const float* __restrict__ src,
                               _Float16* __restrict__ dst, int n)
{
    const int i = blockIdx.x * blockDim.x + threadIdx.x;
    if (i < n) dst[i] = (_Float16)src[i];
}

// ---------------------------------------------------------------------------
// XG[m][g] = x[m,:] @ wih[g,:] + bih[g]   (M=400000, K=64, Nout=192)
// One wave owns 16 rows; A loaded once; each col tile finished (bias+store)
// before moving on -> only one accumulator live, no spills.
// ---------------------------------------------------------------------------
__global__ void __launch_bounds__(256)
xg_gemm_kernel(const float* __restrict__ x,        // (N*T, 64)
               const _Float16* __restrict__ wih,   // (192, 64) f16
               const float* __restrict__ bih,      // (192)
               float* __restrict__ xg)             // (N*T, 192)
{
    const int wave = (int)((blockIdx.x * blockDim.x + threadIdx.x) >> 5);
    if (wave >= (NN * TT) / 16) return;
    const int m_base = wave * 16;
    const int lane = threadIdx.x & 31;
    const int hsel = lane >> 4;
    const int l15  = lane & 15;

    const float* arow = x + (size_t)(m_base + l15) * DD;
    const v16h a0 = load_a_cvt(arow, 0, hsel);
    const v16h a1 = load_a_cvt(arow, 32, hsel);

#pragma unroll
    for (int nt = 0; nt < 12; ++nt) {
        const v16h b0 = load_b_f16(wih, DD, nt * 16, 0, hsel, l15);
        const v16h b1 = load_b_f16(wih, DD, nt * 16, 32, hsel, l15);
        const v8f acc = wmma_k64(a0, a1, b0, b1);

        const int col = nt * 16 + l15;
        const float bv = bih[col];
#pragma unroll
        for (int r = 0; r < 8; ++r) {
            const int row = m_base + r + 8 * hsel;
            xg[(size_t)row * G3 + col] = acc[r] + bv;
        }
    }
}

// ---------------------------------------------------------------------------
// One GRU time step. Wave owns 16 rows of h. Per column-group jt (16 cols),
// compute the r/z/n gate tiles (3 accumulators live), do gate math, store.
// All A (h) reads precede all h writes -> in-place update is safe.
// ---------------------------------------------------------------------------
__global__ void __launch_bounds__(256)
gru_step_kernel(float* __restrict__ h,             // (N, 64) in/out
                const _Float16* __restrict__ whh,  // (192, 64) f16
                const float* __restrict__ bhh,     // (192)
                const float* __restrict__ xg,      // (N*T, 192)
                float* __restrict__ hs,            // (N, T, 64)
                int t)
{
    const int wave = (int)((blockIdx.x * blockDim.x + threadIdx.x) >> 5);
    if (wave >= NN / 16) return;
    const int m_base = wave * 16;
    const int lane = threadIdx.x & 31;
    const int hsel = lane >> 4;
    const int l15  = lane & 15;

    const float* arow = h + (size_t)(m_base + l15) * HH;
    const v16h a0 = load_a_cvt(arow, 0, hsel);
    const v16h a1 = load_a_cvt(arow, 32, hsel);

#pragma unroll
    for (int jt = 0; jt < 4; ++jt) {
        // gate tiles: r -> jt, z -> 4+jt, n -> 8+jt (torch 3H = [r, z, n])
        const v8f accR = wmma_k64(a0, a1,
                                  load_b_f16(whh, HH, (jt)     * 16, 0,  hsel, l15),
                                  load_b_f16(whh, HH, (jt)     * 16, 32, hsel, l15));
        const v8f accZ = wmma_k64(a0, a1,
                                  load_b_f16(whh, HH, (4 + jt) * 16, 0,  hsel, l15),
                                  load_b_f16(whh, HH, (4 + jt) * 16, 32, hsel, l15));
        const v8f accN = wmma_k64(a0, a1,
                                  load_b_f16(whh, HH, (8 + jt) * 16, 0,  hsel, l15),
                                  load_b_f16(whh, HH, (8 + jt) * 16, 32, hsel, l15));

        const int j = jt * 16 + l15;
        const float br = bhh[j];
        const float bz = bhh[HH + j];
        const float bn = bhh[2 * HH + j];
#pragma unroll
        for (int r = 0; r < 8; ++r) {
            const int m = m_base + r + 8 * hsel;
            const float* xrow = xg + ((size_t)m * TT + t) * G3;
            float* hrow = h + (size_t)m * HH;
            float* srow = hs + ((size_t)m * TT + t) * HH;

            const float hr = accR[r] + br;
            const float hz = accZ[r] + bz;
            const float hn = accN[r] + bn;
            const float xr = xrow[j];
            const float xz = xrow[HH + j];
            const float xn = xrow[2 * HH + j];
            const float rg = 1.f / (1.f + __expf(-(xr + hr)));
            const float zg = 1.f / (1.f + __expf(-(xz + hz)));
            const float cg = tanhf(xn + rg * hn);
            const float hv = hrow[j];
            const float hnew = (1.f - zg) * cg + zg * hv;
            hrow[j] = hnew;
            srow[j] = hnew;
        }
    }
}

// ---------------------------------------------------------------------------
// Temporal attention: per (node, h) thread; Linear(T,T) + softmax + weighted sum
// ---------------------------------------------------------------------------
__global__ void attn_t_kernel(const float* __restrict__ hs,  // (N, T, 64)
                              const float* __restrict__ aw,  // (T, T)
                              const float* __restrict__ ab,  // (T)
                              float* __restrict__ embW,      // (N, 3, 64)
                              int w)
{
    const int id = blockIdx.x * blockDim.x + threadIdx.x;
    if (id >= NN * HH) return;
    const int n = id >> 6;
    const int hcol = id & 63;

    float seq[TT];
#pragma unroll
    for (int t = 0; t < TT; ++t)
        seq[t] = hs[((size_t)n * TT + t) * HH + hcol];

    float logits[TT];
    float mx = -3.0e38f;
#pragma unroll
    for (int sp = 0; sp < TT; ++sp) {
        float v = ab[sp];
#pragma unroll
        for (int s = 0; s < TT; ++s) v += seq[s] * aw[sp * TT + s];
        logits[sp] = v;
        mx = v > mx ? v : mx;
    }
    float sum = 0.f;
#pragma unroll
    for (int sp = 0; sp < TT; ++sp) {
        const float e = __expf(logits[sp] - mx);
        logits[sp] = e;
        sum += e;
    }
    float outv = 0.f;
#pragma unroll
    for (int sp = 0; sp < TT; ++sp) outv += (logits[sp] / sum) * seq[sp];

    embW[(size_t)n * (WW * HH) + w * HH + hcol] = outv;
}

// ---------------------------------------------------------------------------
// Weekly attention: Linear(3,3) + softmax over weeks
// ---------------------------------------------------------------------------
__global__ void attn_w_kernel(const float* __restrict__ embW, // (N, 3, 64)
                              const float* __restrict__ ww,   // (3, 3)
                              const float* __restrict__ wb,   // (3)
                              float* __restrict__ weekly)     // (N, 64)
{
    const int id = blockIdx.x * blockDim.x + threadIdx.x;
    if (id >= NN * HH) return;
    const int n = id >> 6;
    const int hcol = id & 63;
    const float s0 = embW[(size_t)n * (WW * HH) + 0 * HH + hcol];
    const float s1 = embW[(size_t)n * (WW * HH) + 1 * HH + hcol];
    const float s2 = embW[(size_t)n * (WW * HH) + 2 * HH + hcol];
    float l[3];
    float mx = -3.0e38f;
#pragma unroll
    for (int k = 0; k < 3; ++k) {
        l[k] = wb[k] + s0 * ww[k * 3 + 0] + s1 * ww[k * 3 + 1] + s2 * ww[k * 3 + 2];
        mx = l[k] > mx ? l[k] : mx;
    }
    float sum = 0.f;
#pragma unroll
    for (int k = 0; k < 3; ++k) { l[k] = __expf(l[k] - mx); sum += l[k]; }
    weekly[id] = (l[0] * s0 + l[1] * s1 + l[2] * s2) / sum;
}

// ---------------------------------------------------------------------------
// GAT projection: xgat = weekly @ gat_w^T (M=20000, K=64, Nout=64).
// One wave owns 16 rows; A loaded once; tile stored as soon as finished.
// ---------------------------------------------------------------------------
__global__ void __launch_bounds__(256)
gat_proj_kernel(const float* __restrict__ weekly,
                const _Float16* __restrict__ gatw, // (64,64) f16
                float* __restrict__ xgat)
{
    const int wave = (int)((blockIdx.x * blockDim.x + threadIdx.x) >> 5);
    if (wave >= NN / 16) return;
    const int m_base = wave * 16;
    const int lane = threadIdx.x & 31;
    const int hsel = lane >> 4;
    const int l15  = lane & 15;

    const float* arow = weekly + (size_t)(m_base + l15) * HH;
    const v16h a0 = load_a_cvt(arow, 0, hsel);
    const v16h a1 = load_a_cvt(arow, 32, hsel);

#pragma unroll
    for (int nt = 0; nt < 4; ++nt) {
        const v16h b0 = load_b_f16(gatw, HH, nt * 16, 0, hsel, l15);
        const v16h b1 = load_b_f16(gatw, HH, nt * 16, 32, hsel, l15);
        const v8f acc = wmma_k64(a0, a1, b0, b1);

        const int col = nt * 16 + l15;
#pragma unroll
        for (int r = 0; r < 8; ++r) {
            const int row = m_base + r + 8 * hsel;
            xgat[(size_t)row * HH + col] = acc[r];
        }
    }
}

// ---------------------------------------------------------------------------
// Per-node prep: s = xgat . att_src, d = xgat . att_dst, init reductions
// ---------------------------------------------------------------------------
__global__ void gat_node_kernel(const float* __restrict__ xgat,
                                const float* __restrict__ asrc,
                                const float* __restrict__ adst,
                                float* __restrict__ sArr, float* __restrict__ dArr,
                                float* __restrict__ amax, float* __restrict__ den,
                                float* __restrict__ cat)
{
    const int n = blockIdx.x * blockDim.x + threadIdx.x;
    if (n >= NN) return;
    float sv = 0.f, dv = 0.f;
    const float* row = xgat + (size_t)n * HH;
#pragma unroll 8
    for (int k = 0; k < HH; ++k) {
        const float v = row[k];
        sv += v * asrc[k];
        dv += v * adst[k];
        cat[(size_t)n * HH + k] = 0.f;
    }
    sArr[n] = sv;
    dArr[n] = dv;
    amax[n] = -3.0e38f;
    den[n]  = 0.f;
}

__device__ inline void atomicMaxF(float* addr, float val)
{
    int* ai = reinterpret_cast<int*>(addr);
    float cur = __int_as_float(__hip_atomic_load(ai, __ATOMIC_RELAXED,
                                                 __HIP_MEMORY_SCOPE_AGENT));
    while (val > cur) {
        const int assumed = __float_as_int(cur);
        const int prev = atomicCAS(ai, assumed, __float_as_int(val));
        if (prev == assumed) break;
        cur = __int_as_float(prev);
    }
}

// edges 0..E-1 from edge_index; edges E..E+N-1 are self-loops
__device__ inline void edge_pair(const int* __restrict__ ei, int e, int& src, int& dst)
{
    if (e < EE) { src = ei[e]; dst = ei[EE + e]; }
    else        { src = e - EE; dst = e - EE; }
}

__global__ void edge_alpha_kernel(const int* __restrict__ ei,
                                  const float* __restrict__ sArr,
                                  const float* __restrict__ dArr,
                                  float* __restrict__ alpha,
                                  float* __restrict__ amax)
{
    const int e = blockIdx.x * blockDim.x + threadIdx.x;
    if (e >= EE + NN) return;
    int src, dst;
    edge_pair(ei, e, src, dst);
    float a = sArr[src] + dArr[dst];
    a = a > 0.f ? a : 0.2f * a;   // leaky_relu, slope 0.2
    alpha[e] = a;
    atomicMaxF(&amax[dst], a);
}

__global__ void edge_exp_kernel(const int* __restrict__ ei,
                                const float* __restrict__ alpha,
                                const float* __restrict__ amax,
                                float* __restrict__ exb,
                                float* __restrict__ den)
{
    const int e = blockIdx.x * blockDim.x + threadIdx.x;
    if (e >= EE + NN) return;
    int src, dst;
    edge_pair(ei, e, src, dst);
    const float ex = __expf(alpha[e] - amax[dst]);
    exb[e] = ex;
    atomicAdd(&den[dst], ex);
}

__global__ void edge_aggr_kernel(const int* __restrict__ ei,
                                 const float* __restrict__ exb,
                                 const float* __restrict__ den,
                                 const float* __restrict__ xgat,
                                 float* __restrict__ cat)
{
    const int id = blockIdx.x * blockDim.x + threadIdx.x;
    const int e = id >> 6;
    const int hcol = id & 63;
    if (e >= EE + NN) return;
    int src, dst;
    edge_pair(ei, e, src, dst);
    const float coef = exb[e] / den[dst];
    atomicAdd(&cat[(size_t)dst * HH + hcol], coef * xgat[(size_t)src * HH + hcol]);
}

// ---------------------------------------------------------------------------
// Fusion: relu([weekly | cat+gat_b] @ fus_w^T + fus_b)
// ---------------------------------------------------------------------------
__global__ void fusion_kernel(const float* __restrict__ weekly,
                              const float* __restrict__ cat,
                              const float* __restrict__ gatb,
                              const float* __restrict__ fw,  // (64, 128)
                              const float* __restrict__ fb,
                              float* __restrict__ fus)
{
    const int id = blockIdx.x * blockDim.x + threadIdx.x;
    if (id >= NN * HH) return;
    const int n = id >> 6;
    const int hcol = id & 63;
    const float* w1 = fw + (size_t)hcol * (2 * HH);
    const float* wk = weekly + (size_t)n * HH;
    const float* ct = cat + (size_t)n * HH;
    float acc = fb[hcol];
#pragma unroll 8
    for (int k = 0; k < HH; ++k) acc += wk[k] * w1[k];
#pragma unroll 8
    for (int k = 0; k < HH; ++k) acc += (ct[k] + gatb[k]) * w1[HH + k];
    fus[id] = acc > 0.f ? acc : 0.f;
}

__global__ void heads_kernel(const float* __restrict__ fus,
                             const float* __restrict__ rw, const float* __restrict__ rb,
                             const float* __restrict__ cw, const float* __restrict__ cb,
                             float* __restrict__ out)
{
    const int n = blockIdx.x * blockDim.x + threadIdx.x;
    if (n >= NN) return;
    float r = rb[0], c = cb[0];
    const float* f = fus + (size_t)n * HH;
#pragma unroll 8
    for (int k = 0; k < HH; ++k) {
        r += f[k] * rw[k];
        c += f[k] * cw[k];
    }
    out[n] = r;
    out[NN + n] = 1.f / (1.f + __expf(-c));
}

__global__ void zero_kernel(float* __restrict__ p, int n)
{
    const int i = blockIdx.x * blockDim.x + threadIdx.x;
    if (i < n) p[i] = 0.f;
}

// ---------------------------------------------------------------------------
// Host launcher
// ---------------------------------------------------------------------------
extern "C" void kernel_launch(void* const* d_in, const int* in_sizes, int n_in,
                              void* d_out, int out_size, void* d_ws, size_t ws_size,
                              hipStream_t stream)
{
    const float* x[3] = { (const float*)d_in[0], (const float*)d_in[1],
                          (const float*)d_in[2] };
    const float* gru_wih = (const float*)d_in[3];   // (3, 192, 64)
    const float* gru_whh = (const float*)d_in[4];   // (3, 192, 64)
    const float* gru_bih = (const float*)d_in[5];   // (3, 192)
    const float* gru_bhh = (const float*)d_in[6];   // (3, 192)
    const float* att_w   = (const float*)d_in[7];   // (3, 20, 20)
    const float* att_b   = (const float*)d_in[8];   // (3, 20)
    const float* ww_w    = (const float*)d_in[9];   // (3, 3)
    const float* ww_b    = (const float*)d_in[10];  // (3)
    const float* gat_w   = (const float*)d_in[11];  // (64, 64)
    const float* gat_src = (const float*)d_in[12];  // (64)
    const float* gat_dst = (const float*)d_in[13];  // (64)
    const float* gat_b   = (const float*)d_in[14];  // (64)
    const float* fus_w   = (const float*)d_in[15];  // (64, 128)
    const float* fus_b   = (const float*)d_in[16];  // (64)
    const float* reg_w   = (const float*)d_in[17];  // (1, 64)
    const float* reg_b   = (const float*)d_in[18];  // (1)
    const float* cls_w   = (const float*)d_in[19];  // (1, 64)
    const float* cls_b   = (const float*)d_in[20];  // (1)
    const int*   eidx    = (const int*)d_in[21];    // (2, E)
    float* out = (float*)d_out;

    // workspace carve-up
    float* ws = (float*)d_ws;
    size_t off = 0;
    auto carve = [&](size_t n) { float* p = ws + off; off += n; return p; };
    float* XG     = carve((size_t)NN * TT * G3);   // 76.8M f32
    float* hbuf   = carve((size_t)NN * HH);
    float* hs     = carve((size_t)NN * TT * HH);   // 25.6M f32
    float* embW   = carve((size_t)NN * WW * HH);
    float* weekly = carve((size_t)NN * HH);
    float* xgat   = carve((size_t)NN * HH);
    float* sArr   = carve(NN);
    float* dArr   = carve(NN);
    float* amax   = carve(NN);
    float* den    = carve(NN);
    float* alpha  = carve(EE + NN);
    float* exb    = carve(EE + NN);
    float* cat    = carve((size_t)NN * HH);
    float* fus    = carve((size_t)NN * HH);
    // f16 weight copies (carved in float units; 2 halves per float)
    _Float16* wih_h  = (_Float16*)carve((size_t)WW * G3 * DD / 2); // 3*192*64 halves
    _Float16* whh_h  = (_Float16*)carve((size_t)WW * G3 * HH / 2);
    _Float16* gatw_h = (_Float16*)carve((size_t)HH * HH / 2);

    const int B = 256;
    const int nhBlocks   = (NN * HH + B - 1) / B;          // 5000
    const int nodeBlocks = (NN + B - 1) / B;               // 79
    const int edgeBlocks = (EE + NN + B - 1) / B;          // 2579
    const int aggrBlocks = ((EE + NN) * 64 + B - 1) / B;   // 165000
    const int xgBlocks   = ((NN * TT / 16) + 7) / 8;       // 3125 (8 waves/blk)
    const int gruBlocks  = ((NN / 16) + 7) / 8;            // 157
    const int projBlocks = ((NN / 16) + 7) / 8;            // 157

    // ---- weight pre-conversion to f16 (tiny, once per call) ----
    cvt_f16_kernel<<<(WW * G3 * DD + B - 1) / B, B, 0, stream>>>(
        gru_wih, wih_h, WW * G3 * DD);
    cvt_f16_kernel<<<(WW * G3 * HH + B - 1) / B, B, 0, stream>>>(
        gru_whh, whh_h, WW * G3 * HH);
    cvt_f16_kernel<<<(HH * HH + B - 1) / B, B, 0, stream>>>(
        gat_w, gatw_h, HH * HH);

    // ---- per-week GRU encoder + temporal attention ----
    for (int w = 0; w < WW; ++w) {
        zero_kernel<<<nhBlocks, B, 0, stream>>>(hbuf, NN * HH);
        xg_gemm_kernel<<<xgBlocks, B, 0, stream>>>(
            x[w], wih_h + (size_t)w * G3 * DD, gru_bih + (size_t)w * G3, XG);
        for (int t = 0; t < TT; ++t) {
            gru_step_kernel<<<gruBlocks, B, 0, stream>>>(
                hbuf, whh_h + (size_t)w * G3 * HH, gru_bhh + (size_t)w * G3,
                XG, hs, t);
        }
        attn_t_kernel<<<nhBlocks, B, 0, stream>>>(
            hs, att_w + (size_t)w * TT * TT, att_b + (size_t)w * TT, embW, w);
    }

    // ---- weekly attention ----
    attn_w_kernel<<<nhBlocks, B, 0, stream>>>(embW, ww_w, ww_b, weekly);

    // ---- GAT ----
    gat_proj_kernel<<<projBlocks, B, 0, stream>>>(weekly, gatw_h, xgat);
    gat_node_kernel<<<nodeBlocks, B, 0, stream>>>(xgat, gat_src, gat_dst,
                                                  sArr, dArr, amax, den, cat);
    edge_alpha_kernel<<<edgeBlocks, B, 0, stream>>>(eidx, sArr, dArr, alpha, amax);
    edge_exp_kernel<<<edgeBlocks, B, 0, stream>>>(eidx, alpha, amax, exb, den);
    edge_aggr_kernel<<<aggrBlocks, B, 0, stream>>>(eidx, exb, den, xgat, cat);

    // ---- fusion + heads ----
    fusion_kernel<<<nhBlocks, B, 0, stream>>>(weekly, cat, gat_b, fus_w, fus_b, fus);
    heads_kernel<<<nodeBlocks, B, 0, stream>>>(fus, reg_w, reg_b, cls_w, cls_b, out);

    (void)in_sizes; (void)n_in; (void)out_size; (void)ws_size;
}
